// BAModule_44066364457375
// MI455X (gfx1250) — matrix operand order repaired and checked by reading, not compiled, not run on hardware
//
#include <hip/hip_runtime.h>
#include <math.h>

// ---------------------------------------------------------------------------
// MI455X (gfx1250) implementation.
// Roofline: ~2.25 TFLOP of GEMM vs ~2 GB HBM traffic -> matrix-core bound.
// Strategy: bf16 WMMA (v_wmma_f32_16x16x32_bf16) with fp32 accumulation for
// every projection/MLP; double-buffered GLOBAL_LOAD_ASYNC_TO_LDS_B128 staging
// (ASYNCcnt) so global->LDS DMA overlaps WMMA; fp32 VALU for the tiny
// sparsemax attention cores and LayerNorms; fused bias/GELU/residual
// epilogues; bf16 shadow copies of every GEMM input to halve re-read traffic.
// ---------------------------------------------------------------------------

typedef __attribute__((ext_vector_type(16))) __bf16 v16bf;
typedef __attribute__((ext_vector_type(8)))  __bf16 v8bf;
typedef __attribute__((ext_vector_type(8)))  float  v8f;

#define LDSP 40  // LDS row stride in bf16 elems (32 + 8 pad; 80B keeps 16B align)

__device__ __forceinline__ float gelu_exact(float x) {
  return 0.5f * x * (1.0f + erff(x * 0.70710678118654752f));
}

// Exact sparsemax threshold via Michelot's finite projection algorithm.
// Result identical to the sort/cumsum reference formulation.
__device__ __forceinline__ float sparsemax_tau(const float* z, int S) {
  float tau = -3.0e38f;
  int cnt = -1;
  for (int it = 0; it < 16; ++it) {
    float s = 0.f; int c = 0;
    for (int i = 0; i < S; ++i)
      if (z[i] > tau) { s += z[i]; c++; }
    float t = (s - 1.0f) / (float)c;
    if (c == cnt) break;  // support stable -> t == tau
    tau = t; cnt = c;
  }
  return tau;
}

// Issue one 16B-per-lane async cache->LDS DMA (no VGPR data path, ASYNCcnt).
__device__ __forceinline__ void async_b128(unsigned ldsAddr, const __bf16* gptr) {
  asm volatile("global_load_async_to_lds_b128 %0, %1, off"
               :: "v"(ldsAddr), "v"((unsigned long long)gptr)
               : "memory");
}

// ---------------------------------------------------------------------------
// WMMA GEMM: C[M,N] = A[M,K](bf16) * W[K,N](bf16, stored transposed [N][K])
//            (+bias) (GELU?) (+resid fp32) -> outF fp32 and/or outB bf16.
// Block: 256 threads (8 waves), macro tile 128x128, wave tile 64x32.
// K stepped by 32 through double-buffered LDS filled by async-to-LDS DMA.
// ---------------------------------------------------------------------------
__global__ __launch_bounds__(256)
void gemm_bf16_wmma(const __bf16* __restrict__ A, const __bf16* __restrict__ Wt,
                    const float* __restrict__ bias, const float* __restrict__ resid,
                    float* __restrict__ outF, __bf16* __restrict__ outB,
                    int M, int N, int K, int aRowsMod, int doGelu) {
  __shared__ __align__(16) __bf16 As[2][128 * LDSP];
  __shared__ __align__(16) __bf16 Bs[2][128 * LDSP];

  const int tid   = threadIdx.x;
  const int lane  = tid & 31;
  const int wave  = tid >> 5;
  const int waveM = wave >> 2;   // 0..1
  const int waveN = wave & 3;    // 0..3
  const int mBase = blockIdx.y * 128;
  const int nBase = blockIdx.x * 128;
  const int lrow   = lane & 15;
  const int hiHalf = lane >> 4;  // 0 or 1

  // Per-thread copy chunks: 512 chunks of 8 bf16 per 128x32 tile, 2 per thread.
  const int ch0 = tid,        ch1 = tid + 256;
  const int row0 = ch0 >> 2,  row1 = ch1 >> 2;        // 0..127
  const int kc0  = (ch0 & 3) << 3, kc1 = (ch1 & 3) << 3;
  int arow0 = mBase + row0, arow1 = mBase + row1;
  if (aRowsMod) { arow0 %= aRowsMod; arow1 %= aRowsMod; }
  const __bf16* aP0 = A  + (size_t)arow0 * K + kc0;
  const __bf16* aP1 = A  + (size_t)arow1 * K + kc1;
  const __bf16* bP0 = Wt + (size_t)(nBase + row0) * K + kc0;
  const __bf16* bP1 = Wt + (size_t)(nBase + row1) * K + kc1;

  v8f acc[4][2] = {};

  auto issue_tile = [&](int buf, int k0) {
    async_b128((unsigned)(uintptr_t)&As[buf][row0 * LDSP + kc0], aP0 + k0);
    async_b128((unsigned)(uintptr_t)&As[buf][row1 * LDSP + kc1], aP1 + k0);
    async_b128((unsigned)(uintptr_t)&Bs[buf][row0 * LDSP + kc0], bP0 + k0);
    async_b128((unsigned)(uintptr_t)&Bs[buf][row1 * LDSP + kc1], bP1 + k0);
  };

  issue_tile(0, 0);                       // prologue: tile 0 in flight
  int cur = 0;

  for (int k0 = 0; k0 < K; k0 += 32) {
    if (k0 + 32 < K) {
      issue_tile(cur ^ 1, k0 + 32);       // DMA next tile while we compute
      asm volatile("s_wait_asynccnt 0x4" ::: "memory");  // tile k0 resident
    } else {
      asm volatile("s_wait_asynccnt 0x0" ::: "memory");
    }
    __syncthreads();                      // all waves' DMA for tile k0 visible

    v16bf af[4], bfv[2];
#pragma unroll
    for (int mt = 0; mt < 4; ++mt) {
      // A 16-bit layout: lanes 0-15: K {0..7,16..23}; lanes 16-31: K {8..15,24..31}
      int r  = waveM * 64 + mt * 16 + lrow;
      int kb = hiHalf ? 8 : 0;
      v8bf lo = *reinterpret_cast<const v8bf*>(&As[cur][r * LDSP + kb]);
      v8bf hi = *reinterpret_cast<const v8bf*>(&As[cur][r * LDSP + kb + 16]);
      af[mt]  = __builtin_shufflevector(lo, hi, 0,1,2,3,4,5,6,7,8,9,10,11,12,13,14,15);
    }
#pragma unroll
    for (int nt = 0; nt < 2; ++nt) {
      // B 16-bit layout: lane = N col; lanes 0-15: K {0..7,16..23}; lanes 16-31: K {8..15,24..31}
      int c  = waveN * 32 + nt * 16 + lrow;
      int ks = hiHalf ? 8 : 0;
      v8bf lo = *reinterpret_cast<const v8bf*>(&Bs[cur][c * LDSP + ks]);
      v8bf hi = *reinterpret_cast<const v8bf*>(&Bs[cur][c * LDSP + ks + 16]);
      bfv[nt] = __builtin_shufflevector(lo, hi, 0,1,2,3,4,5,6,7,8,9,10,11,12,13,14,15);
    }
#pragma unroll
    for (int mt = 0; mt < 4; ++mt)
#pragma unroll
      for (int nt = 0; nt < 2; ++nt)
        acc[mt][nt] = __builtin_amdgcn_wmma_f32_16x16x32_bf16(
            false, af[mt], false, bfv[nt], (short)0, acc[mt][nt], false, false);

    __syncthreads();   // everyone done reading buf `cur` before it is re-filled
    cur ^= 1;
  }

  // Epilogue. C/D layout: VGPR r, lanes 0-15 -> M=r, N=lane; lanes 16-31 -> M=r+8.
  const int rowOff = hiHalf ? 8 : 0;
#pragma unroll
  for (int mt = 0; mt < 4; ++mt) {
#pragma unroll
    for (int nt = 0; nt < 2; ++nt) {
      int col  = nBase + waveN * 32 + nt * 16 + lrow;
      float bv = bias ? bias[col] : 0.0f;
#pragma unroll
      for (int r = 0; r < 8; ++r) {
        int row   = mBase + waveM * 64 + mt * 16 + r + rowOff;
        float v   = acc[mt][nt][r] + bv;
        if (doGelu) v = gelu_exact(v);
        size_t off = (size_t)row * N + col;
        if (resid) v += resid[off];
        if (outF) outF[off] = v;
        if (outB) outB[off] = (__bf16)v;
      }
    }
  }
}

// ---------------------------------------------------------------------------
// feat attention: one wave per (group, head); L = S = 10, DK = 64.
// Output written in "mixed" layout: base + h*L*64 + l*64 + e (bf16).
// ---------------------------------------------------------------------------
__global__ void attn_small_kernel(const float* __restrict__ q, const float* __restrict__ k,
                                  const float* __restrict__ v, __bf16* __restrict__ outB,
                                  int L, float scale) {
  int g = blockIdx.x, h = blockIdx.y;
  __shared__ float sq[10 * 64], sk[10 * 64], sv[10 * 64];
  int n = L * 64;
  size_t base = (size_t)g * L * 512 + (size_t)h * 64;
  for (int i = threadIdx.x; i < n; i += 32) {
    int l = i >> 6, e = i & 63;
    size_t o = base + (size_t)l * 512 + e;
    sq[i] = q[o]; sk[i] = k[o]; sv[i] = v[o];
  }
  __syncthreads();
  int l = threadIdx.x;
  if (l < L) {
    float z[10];
    for (int s = 0; s < L; ++s) {
      float acc = 0.f;
      for (int e = 0; e < 64; ++e) acc += sq[l * 64 + e] * sk[s * 64 + e];
      z[s] = acc * scale;
    }
    float tau = sparsemax_tau(z, L);
    size_t ob = (size_t)g * L * 512 + (size_t)h * L * 64 + (size_t)l * 64;
    for (int e = 0; e < 64; ++e) {
      float acc = 0.f;
      for (int s = 0; s < L; ++s) {
        float p = z[s] - tau;
        if (p > 0.f) acc += p * sv[s * 64 + e];
      }
      outB[ob + e] = (__bf16)acc;
    }
  }
}

// ---------------------------------------------------------------------------
// pool attention: one wave per (group, head, l); L=8 queries, S=256 keys.
// Wave-parallel Michelot sparsemax over 256 scores (8 per lane).
// ---------------------------------------------------------------------------
__global__ void attn_pool_kernel(const float* __restrict__ q, const float* __restrict__ k,
                                 const float* __restrict__ v, __bf16* __restrict__ outB,
                                 float scale) {
  int g = blockIdx.x, h = blockIdx.y, l = blockIdx.z;
  int lane = threadIdx.x;
  __shared__ float sq[64];
  __shared__ float sp[256];
  size_t qoff = ((size_t)g * 8 + l) * 512 + (size_t)h * 64;
  sq[lane] = q[qoff + lane];
  sq[lane + 32] = q[qoff + lane + 32];
  __syncthreads();
  float z[8];
  for (int j = 0; j < 8; ++j) {
    int s = j * 32 + lane;
    size_t koff = ((size_t)g * 256 + s) * 512 + (size_t)h * 64;
    float acc = 0.f;
    for (int e = 0; e < 64; ++e) acc += sq[e] * k[koff + e];
    z[j] = acc * scale;
  }
  float tau = -3.0e38f; int cnt = -1;
  for (int it = 0; it < 300; ++it) {
    float s2 = 0.f; int c2 = 0;
    for (int j = 0; j < 8; ++j) if (z[j] > tau) { s2 += z[j]; c2++; }
    for (int m = 16; m >= 1; m >>= 1) {
      s2 += __shfl_xor(s2, m, 32);
      c2 += __shfl_xor(c2, m, 32);
    }
    float t = (s2 - 1.0f) / (float)c2;
    if (c2 == cnt) break;
    tau = t; cnt = c2;
  }
  for (int j = 0; j < 8; ++j) sp[j * 32 + lane] = fmaxf(z[j] - tau, 0.f);
  __syncthreads();
  float a0 = 0.f, a1 = 0.f;
  for (int s = 0; s < 256; ++s) {
    float p = sp[s];
    if (p > 0.f) {
      size_t voff = ((size_t)g * 256 + s) * 512 + (size_t)h * 64;
      a0 += p * v[voff + lane];
      a1 += p * v[voff + lane + 32];
    }
  }
  size_t ob = (size_t)g * (8 * 512) + (size_t)h * (8 * 64) + (size_t)l * 64;
  outB[ob + lane]      = (__bf16)a0;
  outB[ob + 32 + lane] = (__bf16)a1;
}

// ---------------------------------------------------------------------------
// emb attention: one 256-thread block per (group, head); L=256, S=8.
// ---------------------------------------------------------------------------
__global__ __launch_bounds__(256)
void attn_emb_kernel(const float* __restrict__ q, const float* __restrict__ k,
                     const float* __restrict__ v, __bf16* __restrict__ outB, float scale) {
  int g = blockIdx.x, h = blockIdx.y;
  __shared__ float sk[8 * 64], sv[8 * 64];
  for (int i = threadIdx.x; i < 512; i += 256) {
    int s = i >> 6, e = i & 63;
    size_t o = ((size_t)g * 8 + s) * 512 + (size_t)h * 64 + e;
    sk[i] = k[o]; sv[i] = v[o];
  }
  __syncthreads();
  int l = threadIdx.x;
  size_t qoff = ((size_t)g * 256 + l) * 512 + (size_t)h * 64;
  float z[8];
  for (int s = 0; s < 8; ++s) {
    float acc = 0.f;
    for (int e = 0; e < 64; ++e) acc += q[qoff + e] * sk[s * 64 + e];
    z[s] = acc * scale;
  }
  float tau = sparsemax_tau(z, 8);
  float p[8];
  for (int s = 0; s < 8; ++s) p[s] = fmaxf(z[s] - tau, 0.f);
  size_t ob = (size_t)g * (256 * 512) + (size_t)h * (256 * 64) + (size_t)l * 64;
  for (int e = 0; e < 64; ++e) {
    float acc = 0.f;
    for (int s = 0; s < 8; ++s) acc += p[s] * sv[s * 64 + e];
    outB[ob + e] = (__bf16)acc;
  }
}

// ---------------------------------------------------------------------------
// LayerNorm over D=512, one wave per row. Optional residual input, dual
// fp32/bf16 outputs, and row remap (1: (b,e,p)->(b,p,e), 2: (b,p,e)->(b,e,p)).
// ---------------------------------------------------------------------------
__global__ __launch_bounds__(256)
void ln_kernel(const float* __restrict__ in, const float* __restrict__ res,
               const float* __restrict__ gam, const float* __restrict__ bet,
               float* __restrict__ outF, __bf16* __restrict__ outB,
               int nrows, int remap) {
  int row = blockIdx.x * 8 + (threadIdx.x >> 5);
  if (row >= nrows) return;
  int lane = threadIdx.x & 31;
  const float* src  = in + (size_t)row * 512;
  const float* rsrc = res ? res + (size_t)row * 512 : nullptr;
  float xv[16];
  float s = 0.f;
#pragma unroll
  for (int i = 0; i < 16; ++i) {
    int c = lane + (i << 5);
    float t = src[c];
    if (rsrc) t += rsrc[c];
    xv[i] = t; s += t;
  }
#pragma unroll
  for (int m = 16; m >= 1; m >>= 1) s += __shfl_xor(s, m, 32);
  float mean = s * (1.0f / 512.0f);
  float vs = 0.f;
#pragma unroll
  for (int i = 0; i < 16; ++i) { float d = xv[i] - mean; vs += d * d; }
#pragma unroll
  for (int m = 16; m >= 1; m >>= 1) vs += __shfl_xor(vs, m, 32);
  float rstd = rsqrtf(vs * (1.0f / 512.0f) + 1e-5f);
  int orow = row;
  if (remap == 1) {           // (b*256+e)*10+p  ->  (b*10+p)*256+e
    int p = row % 10; int be = row / 10; int e = be & 255; int b = be >> 8;
    orow = ((b * 10 + p) << 8) + e;
  } else if (remap == 2) {    // (b*10+p)*256+e  ->  (b*256+e)*10+p
    int e = row & 255; int bp = row >> 8; int p = bp % 10; int b = bp / 10;
    orow = (b * 256 + e) * 10 + p;
  }
  size_t ob = (size_t)orow * 512;
#pragma unroll
  for (int i = 0; i < 16; ++i) {
    int c = lane + (i << 5);
    float y = (xv[i] - mean) * rstd * gam[c] + bet[c];
    if (outF) outF[ob + c] = y;
    if (outB) outB[ob + c] = (__bf16)y;
  }
}

// ---------------------------------------------------------------------------
// Conversion helpers.
// ---------------------------------------------------------------------------
__global__ void f2bf_kernel(const float* __restrict__ in, __bf16* __restrict__ out, size_t n) {
  size_t i = (size_t)blockIdx.x * blockDim.x + threadIdx.x;
  size_t stride = (size_t)gridDim.x * blockDim.x;
  for (; i < n; i += stride) out[i] = (__bf16)in[i];
}

// wt[n*K+k] = (bf16) w[k*N+n]   (pre-transpose weights to [N][K])
__global__ void wconvT_kernel(const float* __restrict__ w, __bf16* __restrict__ wt,
                              int K, int N) {
  size_t n = (size_t)K * N;
  size_t i = (size_t)blockIdx.x * blockDim.x + threadIdx.x;
  size_t stride = (size_t)gridDim.x * blockDim.x;
  for (; i < n; i += stride) {
    size_t kk = i % (size_t)K, nn = i / (size_t)K;
    wt[i] = (__bf16)w[kk * (size_t)N + nn];
  }
}

// ---------------------------------------------------------------------------
// Host orchestration.
// ---------------------------------------------------------------------------
extern "C" void kernel_launch(void* const* d_in, const int* in_sizes, int n_in,
                              void* d_out, int out_size, void* d_ws, size_t ws_size,
                              hipStream_t stream) {
  (void)in_sizes; (void)n_in; (void)out_size; (void)ws_size;

  const int T1 = 64 * 256 * 10;           // 163840 feat-path rows
  const size_t N1 = (size_t)T1 * 512;     // 83,886,080
  const int T2 = 640 * 8;                 // 5120 pool rows
  const size_t N2 = (size_t)T2 * 512;

  const float* x       = (const float*)d_in[0];
  const float* pooling = (const float*)d_in[33];
  auto fin = [&](int i) { return (const float*)d_in[i]; };

  // ---- workspace carve (256B aligned) ----
  char* ws = (char*)d_ws;
  size_t off = 0;
  auto carve = [&](size_t bytes) {
    void* p = ws + off;
    off += (bytes + 255) & ~(size_t)255;
    return p;
  };
  float*  F0 = (float*)carve(N1 * 4);
  float*  F1 = (float*)carve(N1 * 4);
  float*  F2 = (float*)carve(N1 * 4);
  __bf16* B0 = (__bf16*)carve(N1 * 2);
  __bf16* B1 = (__bf16*)carve(N1 * 2);
  __bf16* M0 = (__bf16*)carve((size_t)T1 * 2048 * 2);
  __bf16* PB = (__bf16*)carve((size_t)80 * 512 * 2);    // pooling bf16 (80 rows)
  float*  S1 = (float*)carve(N2 * 4);
  float*  S2 = (float*)carve(N2 * 4);
  __bf16* S3 = (__bf16*)carve(N2 * 2);
  __bf16* S5 = (__bf16*)carve(N2 * 2);

  struct WSpec { int idx; int K; int N; };
  const WSpec wspec[16] = {
    {1,512,512},{3,512,512},{5,512,512},{7,512,512},      // feat  wq wk wv wo
    {9,512,512},{11,512,512},{13,512,512},{15,512,512},   // pool  wq wk wv wo
    {17,512,512},{19,512,512},{21,512,512},{23,512,512},  // emb   wq wk wv wo
    {25,512,2048},{27,2048,512},                          // mlp1  w1 w2
    {29,512,2048},{31,2048,512},                          // mlp2  w1 w2
  };
  __bf16* WT[16];
  for (int i = 0; i < 16; ++i)
    WT[i] = (__bf16*)carve((size_t)wspec[i].K * wspec[i].N * 2);

  // ---- weight + input conversions ----
  for (int i = 0; i < 16; ++i) {
    size_t n = (size_t)wspec[i].K * wspec[i].N;
    int blocks = (int)((n + 255) / 256); if (blocks > 4096) blocks = 4096;
    wconvT_kernel<<<blocks, 256, 0, stream>>>(fin(wspec[i].idx), WT[i],
                                              wspec[i].K, wspec[i].N);
  }
  f2bf_kernel<<<4096, 256, 0, stream>>>(x, B0, N1);
  f2bf_kernel<<<64, 256, 0, stream>>>(pooling, PB, (size_t)80 * 512);

  auto gemm = [&](const __bf16* A, const __bf16* Wt, const float* bias,
                  const float* resid, float* oF, __bf16* oB,
                  int M, int N, int K, int mod, int gelu) {
    dim3 g(N / 128, M / 128);
    gemm_bf16_wmma<<<g, 256, 0, stream>>>(A, Wt, bias, resid, oF, oB,
                                          M, N, K, mod, gelu);
  };
  auto ln = [&](const float* in, const float* res, const float* g, const float* b,
                float* oF, __bf16* oB, int rows, int remap) {
    ln_kernel<<<rows / 8, 256, 0, stream>>>(in, res, g, b, oF, oB, rows, remap);
  };

  // ================= Stage A: feature attention over patches ================
  gemm(B0, WT[0], fin(2), nullptr, F0, nullptr, T1, 512, 512, 0, 0);   // qf
  gemm(B0, WT[1], fin(4), nullptr, F1, B1,      T1, 512, 512, 0, 0);   // kf (+bf16)
  gemm(B1, WT[2], fin(6), nullptr, F2, nullptr, T1, 512, 512, 0, 0);   // vf = kf@wv
  attn_small_kernel<<<dim3(16384, 8), 32, 0, stream>>>(F0, F1, F2, B1, 10, 0.125f);
  gemm(B1, WT[3], fin(8), nullptr, F0, nullptr, T1, 512, 512, 0, 0);   // feat_enc
  ln(F0, x, fin(34), fin(35), F2, B0, T1, 0);                          // h1 = LN1(x+enc)
  gemm(B0, WT[12], fin(26), nullptr, nullptr, M0, T1, 2048, 512, 0, 1); // mid1 = GELU
  gemm(M0, WT[13], fin(28), F2, F0, nullptr, T1, 512, 2048, 0, 0);     // feat_out = h1+mlp
  ln(F0, nullptr, fin(36), fin(37), F1, B1, T1, 1);                    // emb_send (remapped)

  // ================= Stage B: pooling attention (q=pooling, kv=emb_send) ===
  gemm(PB, WT[4], fin(10), nullptr, S1, nullptr, T2, 512, 512, 80, 0); // q2 (tiled pooling)
  gemm(B1, WT[5], fin(12), nullptr, F0, B0,      T1, 512, 512, 0, 0);  // k2 (+bf16)
  gemm(B0, WT[6], fin(14), nullptr, F2, nullptr, T1, 512, 512, 0, 0);  // v2 = k2@wv
  attn_pool_kernel<<<dim3(640, 8, 8), 32, 0, stream>>>(S1, F0, F2, S3, 0.125f);
  gemm(S3, WT[7], fin(16), nullptr, nullptr, S5, T2, 512, 512, 0, 0);  // emb_buffer (bf16)

  // ================= Stage C: emb attention (q=emb_send, kv=emb_buffer) ====
  gemm(B1, WT[8],  fin(18), nullptr, F0, nullptr, T1, 512, 512, 0, 0); // q3
  gemm(S5, WT[9],  fin(20), nullptr, S1, S3,      T2, 512, 512, 0, 0); // k3 (+bf16)
  gemm(S3, WT[10], fin(22), nullptr, S2, nullptr, T2, 512, 512, 0, 0); // v3 = k3@wv
  attn_emb_kernel<<<dim3(640, 8), 256, 0, stream>>>(F0, S1, S2, B0, 0.125f);
  gemm(B0, WT[11], fin(24), F1, F2, nullptr, T1, 512, 512, 0, 0);      // emb_send + receive
  ln(F2, nullptr, fin(38), fin(39), F0, B0, T1, 0);                    // emb_out = LN3
  gemm(B0, WT[14], fin(30), nullptr, nullptr, M0, T1, 2048, 512, 0, 1); // mid2 = GELU
  gemm(M0, WT[15], fin(32), F0, F2, nullptr, T1, 512, 2048, 0, 0);     // emb_out + mlp2
  ln(F2, nullptr, fin(40), fin(41), (float*)d_out, nullptr, T1, 2);    // LN4 -> output (remapped)
}
